// MultiHeadAttention_12713103196556
// MI455X (gfx1250) — matrix operand order repaired and checked by reading, not compiled, not run on hardware
//
#include <hip/hip_runtime.h>
#include <hip/hip_bf16.h>

// ---- problem constants (from reference) ----
constexpr int kB = 2;
constexpr int kT = 2048;
constexpr int kC = 1024;
constexpr int kH = 16;
constexpr int kD = 64;
constexpr int kM = kB * kT;       // 4096 rows for projection GEMMs

typedef __attribute__((ext_vector_type(16))) __bf16 v16bf;
typedef __attribute__((ext_vector_type(8)))  __bf16 v8bf;
typedef __attribute__((ext_vector_type(8)))  float  v8f;

__device__ __forceinline__ unsigned short f2bf(float f) {
  __bf16 h = (__bf16)f;
  return __builtin_bit_cast(unsigned short, h);
}

// ---------------------------------------------------------------------------
// WMMA fragment loaders, built from the CDNA5 ISA VGPR layouts (ISA 7.12.2).
//
// A-matrix 16x32 bf16: lane m = lane&15 holds row m.
//   lanes 0-15 : K = {0..7} in elems 0..7,  {16..23} in elems 8..15
//   lanes 16-31: K = {8..15} in elems 0..7, {24..31} in elems 8..15
// -> per lane: two contiguous 16-byte loads at k0 + hi*8 and k0 + 16 + hi*8.
// ---------------------------------------------------------------------------
__device__ __forceinline__ v16bf load_a16x32(const unsigned short* pu, int ld) {
  const __bf16* p = (const __bf16*)pu;
  const int lane = (int)(threadIdx.x & 31);
  const int m  = lane & 15;
  const int hi = lane >> 4;
  const __bf16* r = p + (size_t)m * ld + hi * 8;
  v8bf lo = *(const v8bf*)(r);
  v8bf hh = *(const v8bf*)(r + 16);
  v16bf out;
#pragma unroll
  for (int i = 0; i < 8; ++i) { out[i] = lo[i]; out[i + 8] = hh[i]; }
  return out;
}

// ---------------------------------------------------------------------------
// B-matrix 32x16 bf16 where memory holds M2[n][k] row-major (B = M2^T):
//   lane n = lane&15 is column n; same K order as the A loader.
// -> per lane: two 16-byte loads at row (n0+n), offsets k0 + (lane>>4)*8 and +16.
// Caller passes p = base + n0*ld + k0.
// ---------------------------------------------------------------------------
__device__ __forceinline__ v16bf load_b32x16_nk(const unsigned short* pu, int ld) {
  const __bf16* p = (const __bf16*)pu;
  const int lane = (int)(threadIdx.x & 31);
  const int n  = lane & 15;
  const int kb = (lane >> 4) * 8;
  const __bf16* r = p + (size_t)n * ld + kb;
  v8bf lo = *(const v8bf*)(r);
  v8bf hh = *(const v8bf*)(r + 16);
  v16bf out;
#pragma unroll
  for (int i = 0; i < 8; ++i) { out[i] = lo[i]; out[i + 8] = hh[i]; }
  return out;
}

__device__ __forceinline__ v8f wmma_bf16(v16bf a, v16bf b, v8f c) {
  // D = A(16x32) * B(32x16) + C, fp32 accumulate
  return __builtin_amdgcn_wmma_f32_16x16x32_bf16(
      false, a, false, b, (short)0, c, false, false);
}

// ---------------------------------------------------------------------------
// fp32 -> bf16 conversion (x4 per thread)
// ---------------------------------------------------------------------------
__global__ __launch_bounds__(256) void cvt_f32_bf16_x4(
    const float* __restrict__ in, unsigned short* __restrict__ out, int n4) {
  int i = blockIdx.x * 256 + threadIdx.x;
  if (i < n4) {
    float4 f = ((const float4*)in)[i];
    ushort4 o;
    o.x = f2bf(f.x); o.y = f2bf(f.y); o.z = f2bf(f.z); o.w = f2bf(f.w);
    ((ushort4*)out)[i] = o;
  }
}

// ---------------------------------------------------------------------------
// Projection GEMM: Y = X(bf16,[M,C]) * W(bf16,[C,C], torch Linear: W[o][c])^T + b
// Block = 4 waves; each wave owns a 32x64 output tile. 8 WMMA per K-step.
// transpose_out == 0: Y row-major bf16 [M, C]     (used for Q, K)
// transpose_out == 1: Y = [B][H][D][T] bf16       (V transposed for P@V)
// ---------------------------------------------------------------------------
__global__ __launch_bounds__(128) void gemm_proj_bf16(
    const unsigned short* __restrict__ X, const unsigned short* __restrict__ W,
    const float* __restrict__ bias, unsigned short* __restrict__ Y,
    int transpose_out) {
  const int wave = (int)(threadIdx.x >> 5);
  const int lane = (int)(threadIdx.x & 31);
  const int nlo = lane & 15, hi = lane >> 4;
  const int n0 = blockIdx.x * 64;
  const int m0 = blockIdx.y * 128 + wave * 32;

  v8f acc[2][4];
#pragma unroll
  for (int mt = 0; mt < 2; ++mt)
#pragma unroll
    for (int nt = 0; nt < 4; ++nt)
#pragma unroll
      for (int r = 0; r < 8; ++r) acc[mt][nt][r] = 0.0f;

  float bv[4];
#pragma unroll
  for (int nt = 0; nt < 4; ++nt) bv[nt] = bias[n0 + nt * 16 + nlo];

  for (int k0 = 0; k0 < kC; k0 += 32) {
    __builtin_prefetch(X + (size_t)m0 * kC + k0 + 128, 0, 1);
    v16bf a0 = load_a16x32(X + (size_t)(m0)      * kC + k0, kC);
    v16bf a1 = load_a16x32(X + (size_t)(m0 + 16) * kC + k0, kC);
#pragma unroll
    for (int nt = 0; nt < 4; ++nt) {
      v16bf b = load_b32x16_nk(W + (size_t)(n0 + nt * 16) * kC + k0, kC);
      acc[0][nt] = wmma_bf16(a0, b, acc[0][nt]);
      acc[1][nt] = wmma_bf16(a1, b, acc[1][nt]);
    }
  }

#pragma unroll
  for (int mt = 0; mt < 2; ++mt)
#pragma unroll
    for (int nt = 0; nt < 4; ++nt)
#pragma unroll
      for (int r = 0; r < 8; ++r) {
        int row = m0 + mt * 16 + r + 8 * hi;        // token row in [0, B*T)
        int col = n0 + nt * 16 + nlo;               // output channel
        unsigned short q = f2bf(acc[mt][nt][r] + bv[nt]);
        if (!transpose_out) {
          Y[(size_t)row * kC + col] = q;
        } else {
          int b = row / kT, t = row % kT;
          int h = col >> 6, d = col & 63;
          Y[(((size_t)b * kH + h) * kD + d) * kT + t] = q;
        }
      }
}

// ---------------------------------------------------------------------------
// Final projection: out(fp32) = A(bf16,[M,C]) * Wo^T + bo
// ---------------------------------------------------------------------------
__global__ __launch_bounds__(128) void gemm_out_f32(
    const unsigned short* __restrict__ X, const unsigned short* __restrict__ W,
    const float* __restrict__ bias, float* __restrict__ Y) {
  const int wave = (int)(threadIdx.x >> 5);
  const int lane = (int)(threadIdx.x & 31);
  const int nlo = lane & 15, hi = lane >> 4;
  const int n0 = blockIdx.x * 64;
  const int m0 = blockIdx.y * 128 + wave * 32;

  v8f acc[2][4];
#pragma unroll
  for (int mt = 0; mt < 2; ++mt)
#pragma unroll
    for (int nt = 0; nt < 4; ++nt)
#pragma unroll
      for (int r = 0; r < 8; ++r) acc[mt][nt][r] = 0.0f;

  float bv[4];
#pragma unroll
  for (int nt = 0; nt < 4; ++nt) bv[nt] = bias[n0 + nt * 16 + nlo];

  for (int k0 = 0; k0 < kC; k0 += 32) {
    __builtin_prefetch(X + (size_t)m0 * kC + k0 + 128, 0, 1);
    v16bf a0 = load_a16x32(X + (size_t)(m0)      * kC + k0, kC);
    v16bf a1 = load_a16x32(X + (size_t)(m0 + 16) * kC + k0, kC);
#pragma unroll
    for (int nt = 0; nt < 4; ++nt) {
      v16bf b = load_b32x16_nk(W + (size_t)(n0 + nt * 16) * kC + k0, kC);
      acc[0][nt] = wmma_bf16(a0, b, acc[0][nt]);
      acc[1][nt] = wmma_bf16(a1, b, acc[1][nt]);
    }
  }

#pragma unroll
  for (int mt = 0; mt < 2; ++mt)
#pragma unroll
    for (int nt = 0; nt < 4; ++nt)
#pragma unroll
      for (int r = 0; r < 8; ++r) {
        int row = m0 + mt * 16 + r + 8 * hi;
        int col = n0 + nt * 16 + nlo;
        Y[(size_t)row * kC + col] = acc[mt][nt][r] + bv[nt];
      }
}

// ---------------------------------------------------------------------------
// Flash attention, causal. One wave per 16 query rows; 4 waves per block.
// QH, KH: bf16 [B, T, C] (head h = column slice h*64..h*64+63)
// VT:     bf16 [B, H, D, T]  (transposed V)
// AO:     bf16 [B, T, C]     (attention output, pre-projection)
// Online softmax (flash). K-blocks of 32: 4 WMMA for S, 4 WMMA for P@V.
// P is reshaped C-layout -> A-layout through a private 1KB LDS tile
// (DS ops are in-order within a wave; asm memory barrier stops reordering).
// ---------------------------------------------------------------------------
__global__ __launch_bounds__(128) void flash_attn_bf16(
    const unsigned short* __restrict__ QH, const unsigned short* __restrict__ KH,
    const unsigned short* __restrict__ VT, unsigned short* __restrict__ AO) {
  __shared__ unsigned short pbuf[4][16 * 32];   // per-wave P tile, 1KB each

  const int wave = (int)(threadIdx.x >> 5);
  const int lane = (int)(threadIdx.x & 31);
  const int nlo = lane & 15, hi = lane >> 4;
  const int h = blockIdx.y;
  const int b = blockIdx.z;
  const int q0 = blockIdx.x * 64 + wave * 16;

  const unsigned short* Qb = QH + (size_t)b * kT * kC + h * kD;
  const unsigned short* Kb = KH + (size_t)b * kT * kC + h * kD;
  const unsigned short* Vb = VT + ((size_t)b * kH + h) * kD * kT;
  unsigned short*       Ob = AO + (size_t)b * kT * kC + h * kD;

  // Q fragments for this 16-row block, resident for the whole loop
  v16bf qa0 = load_a16x32(Qb + (size_t)q0 * kC + 0,  kC);
  v16bf qa1 = load_a16x32(Qb + (size_t)q0 * kC + 32, kC);

  v8f acc[4];                 // O accumulator, 16 x 64
#pragma unroll
  for (int nt = 0; nt < 4; ++nt)
#pragma unroll
    for (int r = 0; r < 8; ++r) acc[nt][r] = 0.0f;

  float mrow[8], lrow[8];
#pragma unroll
  for (int r = 0; r < 8; ++r) { mrow[r] = -3.0e38f; lrow[r] = 0.0f; }

  const float scale = 0.125f;                    // 1/sqrt(D=64)
  const int nblk = (q0 + 16 + 31) >> 5;          // causal trip count
  unsigned short* pL = pbuf[wave];

  for (int blk = 0; blk < nblk; ++blk) {
    const int k0 = blk * 32;

    // S = Q @ K^T for 16 q-rows x 32 k-cols (two 16-col tiles)
    v8f s0, s1;
#pragma unroll
    for (int r = 0; r < 8; ++r) { s0[r] = 0.0f; s1[r] = 0.0f; }
    s0 = wmma_bf16(qa0, load_b32x16_nk(Kb + (size_t)(k0)      * kC + 0,  kC), s0);
    s0 = wmma_bf16(qa1, load_b32x16_nk(Kb + (size_t)(k0)      * kC + 32, kC), s0);
    s1 = wmma_bf16(qa0, load_b32x16_nk(Kb + (size_t)(k0 + 16) * kC + 0,  kC), s1);
    s1 = wmma_bf16(qa1, load_b32x16_nk(Kb + (size_t)(k0 + 16) * kC + 32, kC), s1);

    const bool need_mask = (k0 + 31 > q0);       // wave-uniform

#pragma unroll
    for (int r = 0; r < 8; ++r) {
      float x0 = s0[r] * scale;
      float x1 = s1[r] * scale;
      if (need_mask) {
        int qrow = q0 + r + 8 * hi;
        x0 = (k0 + nlo      > qrow) ? -1.0e30f : x0;
        x1 = (k0 + 16 + nlo > qrow) ? -1.0e30f : x1;
      }
      // row max over this block's 32 columns (16 lanes per half, 2 tiles)
      float t = fmaxf(x0, x1);
      t = fmaxf(t, __shfl_xor(t, 1, 32));
      t = fmaxf(t, __shfl_xor(t, 2, 32));
      t = fmaxf(t, __shfl_xor(t, 4, 32));
      t = fmaxf(t, __shfl_xor(t, 8, 32));
      float mnew  = fmaxf(mrow[r], t);
      float alpha = __expf(mrow[r] - mnew);
      float p0 = __expf(x0 - mnew);
      float p1 = __expf(x1 - mnew);
      float rs = p0 + p1;
      rs += __shfl_xor(rs, 1, 32);
      rs += __shfl_xor(rs, 2, 32);
      rs += __shfl_xor(rs, 4, 32);
      rs += __shfl_xor(rs, 8, 32);
      lrow[r] = lrow[r] * alpha + rs;
      mrow[r] = mnew;
#pragma unroll
      for (int nt = 0; nt < 4; ++nt) acc[nt][r] *= alpha;

      const int rowl = r + 8 * hi;               // local q row 0..15
      pL[rowl * 32 + nlo]      = f2bf(p0);
      pL[rowl * 32 + 16 + nlo] = f2bf(p1);
    }

    // LDS round-trip: C-layout -> A-layout. Same-wave DS ops are in-order;
    // the barrier only stops the compiler from reordering the b128 loads.
    asm volatile("" ::: "memory");
    v16bf pa = load_a16x32(pL, 32);
    asm volatile("" ::: "memory");

    // O += P @ V   (V pre-transposed: [D, T] row-major per (b,h))
#pragma unroll
    for (int nt = 0; nt < 4; ++nt) {
      v16bf bvf = load_b32x16_nk(Vb + (size_t)(nt * 16) * kT + k0, kT);
      acc[nt] = wmma_bf16(pa, bvf, acc[nt]);
    }
  }

  // epilogue: divide by row sums, store bf16 [B, T, C]
#pragma unroll
  for (int r = 0; r < 8; ++r) {
    float inv = 1.0f / lrow[r];
    int row = q0 + r + 8 * hi;
#pragma unroll
    for (int nt = 0; nt < 4; ++nt) {
      Ob[(size_t)row * kC + nt * 16 + nlo] = f2bf(acc[nt][r] * inv);
    }
  }
}

// ---------------------------------------------------------------------------
// Host launcher
// ---------------------------------------------------------------------------
extern "C" void kernel_launch(void* const* d_in, const int* in_sizes, int n_in,
                              void* d_out, int out_size, void* d_ws, size_t ws_size,
                              hipStream_t stream) {
  (void)in_sizes; (void)n_in; (void)out_size; (void)ws_size;

  const float* q  = (const float*)d_in[0];
  const float* k  = (const float*)d_in[1];
  const float* v  = (const float*)d_in[2];
  // d_in[3] = mask: causal, handled analytically in flash_attn_bf16
  const float* Wq = (const float*)d_in[4];
  const float* bq = (const float*)d_in[5];
  const float* Wk = (const float*)d_in[6];
  const float* bk = (const float*)d_in[7];
  const float* Wv = (const float*)d_in[8];
  const float* bv = (const float*)d_in[9];
  const float* Wo = (const float*)d_in[10];
  const float* bo = (const float*)d_in[11];
  float* out = (float*)d_out;

  const size_t NE = (size_t)kM * kC;   // 4 Mi elements (activations)
  const size_t WE = (size_t)kC * kC;   // 1 Mi elements (weights)

  unsigned short* wsu = (unsigned short*)d_ws;
  unsigned short* xq  = wsu;           // bf16 copies of inputs
  unsigned short* xk  = xq  + NE;
  unsigned short* xv  = xk  + NE;
  unsigned short* wqb = xv  + NE;      // bf16 weights
  unsigned short* wkb = wqb + WE;
  unsigned short* wvb = wkb + WE;
  unsigned short* wob = wvb + WE;
  unsigned short* qh  = wob + WE;      // Q projected   [B,T,C]
  unsigned short* kh  = qh  + NE;      // K projected   [B,T,C]
  unsigned short* vt  = kh  + NE;      // V projected^T [B,H,D,T]
  unsigned short* at  = vt  + NE;      // attention out [B,T,C]
  // total: 7*NE + 4*WE = 32 Mi bf16 = 64 MiB of workspace

  // 1) convert everything to bf16 once
  cvt_f32_bf16_x4<<<(int)(NE / 4 / 256), 256, 0, stream>>>(q,  xq,  (int)(NE / 4));
  cvt_f32_bf16_x4<<<(int)(NE / 4 / 256), 256, 0, stream>>>(k,  xk,  (int)(NE / 4));
  cvt_f32_bf16_x4<<<(int)(NE / 4 / 256), 256, 0, stream>>>(v,  xv,  (int)(NE / 4));
  cvt_f32_bf16_x4<<<(int)(WE / 4 / 256), 256, 0, stream>>>(Wq, wqb, (int)(WE / 4));
  cvt_f32_bf16_x4<<<(int)(WE / 4 / 256), 256, 0, stream>>>(Wk, wkb, (int)(WE / 4));
  cvt_f32_bf16_x4<<<(int)(WE / 4 / 256), 256, 0, stream>>>(Wv, wvb, (int)(WE / 4));
  cvt_f32_bf16_x4<<<(int)(WE / 4 / 256), 256, 0, stream>>>(Wo, wob, (int)(WE / 4));

  // 2) Q/K/V projections (WMMA bf16, fp32 accumulate)
  dim3 gb(128);
  dim3 gg(kC / 64, kM / 128);
  gemm_proj_bf16<<<gg, gb, 0, stream>>>(xq, wqb, bq, qh, 0);
  gemm_proj_bf16<<<gg, gb, 0, stream>>>(xk, wkb, bk, kh, 0);
  gemm_proj_bf16<<<gg, gb, 0, stream>>>(xv, wvb, bv, vt, 1);

  // 3) causal flash attention
  dim3 ag(kT / 64, kH, kB);
  flash_attn_bf16<<<ag, gb, 0, stream>>>(qh, kh, vt, at);

  // 4) output projection -> fp32
  gemm_out_f32<<<gg, gb, 0, stream>>>(at, wob, bo, out);
}
